// LMMDLoss_83889301226267
// MI455X (gfx1250) — matrix operand order repaired and checked by reading, not compiled, not run on hardware
//
#include <hip/hip_runtime.h>
#include <hip/hip_bf16.h>

typedef __attribute__((ext_vector_type(16))) __bf16 v16bf;
typedef __attribute__((ext_vector_type(8)))  __bf16 bf16x8;
typedef __attribute__((ext_vector_type(8)))  float  v8f;

#define BD   2048
#define DD   1024
#define NN   4096
#define NC   12

// ---------------------------------------------------------------------------
// Kernel 1: class-weight prep. One block, deterministic LDS tree reduction.
// sscale[c] = common[c] && s_sum[c]>0 ? 1/s_sum[c] : 0  (same for t).
// ---------------------------------------------------------------------------
__global__ __launch_bounds__(256)
void prep_kernel(const float* __restrict__ src_label,
                 const float* __restrict__ tgt_logits,
                 float* __restrict__ sscale, float* __restrict__ tscale)
{
    __shared__ float red[256][24];
    __shared__ unsigned pres[2];
    const int tid = threadIdx.x;
    if (tid < 2) pres[tid] = 0u;

    float loc[24];
    #pragma unroll
    for (int c = 0; c < 24; ++c) loc[c] = 0.f;
    unsigned lsp = 0u, ltp = 0u;

    for (int r = tid; r < BD; r += 256) {
        const float* s = src_label  + (size_t)r * NC;
        const float* t = tgt_logits + (size_t)r * NC;
        float bs = s[0]; int bis = 0;
        float bt = t[0]; int bit_ = 0;
        #pragma unroll
        for (int c = 0; c < NC; ++c) {
            float vs = s[c]; loc[c]      += vs; if (vs > bs) { bs = vs; bis = c; }
            float vt = t[c]; loc[12 + c] += vt; if (vt > bt) { bt = vt; bit_ = c; }
        }
        lsp |= 1u << bis;
        ltp |= 1u << bit_;
    }
    #pragma unroll
    for (int c = 0; c < 24; ++c) red[tid][c] = loc[c];
    atomicOr(&pres[0], lsp);   // bitwise OR: order-independent, deterministic
    atomicOr(&pres[1], ltp);
    __syncthreads();
    for (int s = 128; s > 0; s >>= 1) {
        if (tid < s)
            #pragma unroll
            for (int c = 0; c < 24; ++c) red[tid][c] += red[tid + s][c];
        __syncthreads();
    }
    if (tid < NC) {
        float ss = red[0][tid], ts = red[0][12 + tid];
        bool common = ((pres[0] >> tid) & 1u) && ((pres[1] >> tid) & 1u);
        sscale[tid] = (common && ss > 0.f) ? 1.f / ss : 0.f;
        tscale[tid] = (common && ts > 0.f) ? 1.f / ts : 0.f;
    }
}

// ---------------------------------------------------------------------------
// Kernel 2: per-row squared norm (exact f32) + split-bf16 (hi/lo) encode of T.
// ---------------------------------------------------------------------------
__global__ __launch_bounds__(256)
void rows_kernel(const float* __restrict__ src, const float* __restrict__ tgt,
                 __bf16* __restrict__ Thi, __bf16* __restrict__ Tlo,
                 float* __restrict__ sq)
{
    __shared__ float red[256];
    const int r   = blockIdx.x;
    const int tid = threadIdx.x;
    const float* row = (r < BD) ? src + (size_t)r * DD
                                : tgt + (size_t)(r - BD) * DD;
    float s = 0.f;
    for (int c = tid; c < DD; c += 256) {
        float x = row[c];
        __bf16 h = (__bf16)x;
        float hf = (float)h;
        __bf16 l = (__bf16)(x - hf);
        Thi[(size_t)r * DD + c] = h;
        Tlo[(size_t)r * DD + c] = l;
        s += x * x;
    }
    red[tid] = s;
    __syncthreads();
    for (int k = 128; k > 0; k >>= 1) {
        if (tid < k) red[tid] += red[tid + k];
        __syncthreads();
    }
    if (tid == 0) sq[r] = red[0];
}

// ---------------------------------------------------------------------------
// Fragment loaders (wave32) from LDS row-major tiles; contiguous K order, same for A and B.
// A (16x32 bf16): lane holds row M=lane&15, elems = 16 consecutive K at 16*hl.
// B (32x16 bf16): lane holds col N=lane&15, elems = 16 consecutive K at 16*hl.
// ---------------------------------------------------------------------------
__device__ inline v16bf ld_fragA(const __bf16* rowp, int hl)
{
    bf16x8 a0 = *(const bf16x8*)(rowp + hl * 16);
    bf16x8 a1 = *(const bf16x8*)(rowp + hl * 16 + 8);
    return __builtin_shufflevector(a0, a1, 0,1,2,3,4,5,6,7,8,9,10,11,12,13,14,15);
}
__device__ inline v16bf ld_fragB(const __bf16* rowp, int hl)
{
    bf16x8 b0 = *(const bf16x8*)(rowp + hl * 16);
    bf16x8 b1 = *(const bf16x8*)(rowp + hl * 16 + 8);
    return __builtin_shufflevector(b0, b1, 0,1,2,3,4,5,6,7,8,9,10,11,12,13,14,15);
}

// ---------------------------------------------------------------------------
// Kernel 3: Gram = T*T^T via split-bf16 WMMA.
// Block tile 128x128, 4 waves (2x2), wave tile 64x64 = 4x4 C-frags.
// Per wave per k-step: 32 ds_load_b128 fragment loads vs 48 v_wmma -> matrix
// ops dominate issue. global_prefetch of the next k-slice hides staging.
// ---------------------------------------------------------------------------
__global__ __launch_bounds__(128)
void gram_kernel(const __bf16* __restrict__ Thi, const __bf16* __restrict__ Tlo,
                 const float* __restrict__ sq, float* __restrict__ gram,
                 float* __restrict__ l2part)
{
    __shared__ __align__(16) __bf16 sIhi[128 * 32], sIlo[128 * 32];
    __shared__ __align__(16) __bf16 sJhi[128 * 32], sJlo[128 * 32];
    __shared__ float red[128];

    const int tid  = threadIdx.x;
    const int lane = tid & 31;
    const int wave = tid >> 5;         // 0..3
    const int hl   = lane >> 4;        // half-wave select
    const int mrow = lane & 15;
    const int Mw   = (wave >> 1) * 64; // wave tile origin in block (M)
    const int Nw   = (wave & 1)  * 64; // (N)
    const int rowI = blockIdx.y * 128; // Gram rows
    const int rowJ = blockIdx.x * 128; // Gram cols

    v8f acc[4][4];
    #pragma unroll
    for (int mi = 0; mi < 4; ++mi)
        #pragma unroll
        for (int ni = 0; ni < 4; ++ni)
            #pragma unroll
            for (int r = 0; r < 8; ++r) acc[mi][ni][r] = 0.f;

    const int lr   = tid >> 1;         // 0..63 : rows lr and lr+64
    const int lseg = (tid & 1) * 16;   // 0/16  : 16-element segment

    for (int kb = 0; kb < DD; kb += 32) {
        __syncthreads();
        #pragma unroll
        for (int h = 0; h < 2; ++h) {
            const int rr = lr + h * 64;
            const size_t gi = (size_t)(rowI + rr) * DD + kb + lseg;
            const size_t gj = (size_t)(rowJ + rr) * DD + kb + lseg;
            const int so = rr * 32 + lseg;
            *(bf16x8*)(sIhi + so)     = *(const bf16x8*)(Thi + gi);
            *(bf16x8*)(sIhi + so + 8) = *(const bf16x8*)(Thi + gi + 8);
            *(bf16x8*)(sIlo + so)     = *(const bf16x8*)(Tlo + gi);
            *(bf16x8*)(sIlo + so + 8) = *(const bf16x8*)(Tlo + gi + 8);
            *(bf16x8*)(sJhi + so)     = *(const bf16x8*)(Thi + gj);
            *(bf16x8*)(sJhi + so + 8) = *(const bf16x8*)(Thi + gj + 8);
            *(bf16x8*)(sJlo + so)     = *(const bf16x8*)(Tlo + gj);
            *(bf16x8*)(sJlo + so + 8) = *(const bf16x8*)(Tlo + gj + 8);
            // Prefetch next k-slice toward cache while this slice computes.
            if (kb + 32 < DD) {
                __builtin_prefetch((const void*)(Thi + gi + 32), 0, 3);
                __builtin_prefetch((const void*)(Tlo + gi + 32), 0, 3);
                __builtin_prefetch((const void*)(Thi + gj + 32), 0, 3);
                __builtin_prefetch((const void*)(Tlo + gj + 32), 0, 3);
            }
        }
        __syncthreads();

        v16bf ahi[4], alo[4];
        #pragma unroll
        for (int mi = 0; mi < 4; ++mi) {
            const int r0 = (Mw + mi * 16 + mrow) * 32;
            ahi[mi] = ld_fragA(sIhi + r0, hl);
            alo[mi] = ld_fragA(sIlo + r0, hl);
        }
        #pragma unroll
        for (int ni = 0; ni < 4; ++ni) {
            const int r0 = (Nw + ni * 16 + mrow) * 32;
            v16bf bhi = ld_fragB(sJhi + r0, hl);
            v16bf blo = ld_fragB(sJlo + r0, hl);
            #pragma unroll
            for (int mi = 0; mi < 4; ++mi) {
                acc[mi][ni] = __builtin_amdgcn_wmma_f32_16x16x32_bf16(
                    false, ahi[mi], false, bhi, (short)0, acc[mi][ni], false, false);
                acc[mi][ni] = __builtin_amdgcn_wmma_f32_16x16x32_bf16(
                    false, ahi[mi], false, blo, (short)0, acc[mi][ni], false, false);
                acc[mi][ni] = __builtin_amdgcn_wmma_f32_16x16x32_bf16(
                    false, alo[mi], false, bhi, (short)0, acc[mi][ni], false, false);
            }
        }
    }

    // Epilogue: store Gram tile + accumulate clamped pairwise-L2 (for bandwidth)
    float lsum = 0.f;
    #pragma unroll
    for (int mi = 0; mi < 4; ++mi)
        #pragma unroll
        for (int ni = 0; ni < 4; ++ni) {
            const int gm = rowI + Mw + mi * 16 + hl * 8;  // C frag: M = r + 8*hl
            const int gn = rowJ + Nw + ni * 16 + mrow;    //         N = lane&15
            const float sqn = sq[gn];
            #pragma unroll
            for (int r = 0; r < 8; ++r) {
                const float c = acc[mi][ni][r];
                gram[(size_t)(gm + r) * NN + gn] = c;
                lsum += fmaxf(sq[gm + r] + sqn - 2.f * c, 0.f);
            }
        }
    red[tid] = lsum;
    __syncthreads();
    for (int k = 64; k > 0; k >>= 1) {
        if (tid < k) red[tid] += red[tid + k];
        __syncthreads();
    }
    if (tid == 0) l2part[blockIdx.y * 32 + blockIdx.x] = red[0];
}

// ---------------------------------------------------------------------------
// Fixed-order single-block reduction (deterministic).
// ---------------------------------------------------------------------------
__global__ __launch_bounds__(256)
void reduce_kernel(const float* __restrict__ in, int n, float* __restrict__ out)
{
    __shared__ float red[256];
    const int tid = threadIdx.x;
    float s = 0.f;
    for (int k = tid; k < n; k += 256) s += in[k];
    red[tid] = s;
    __syncthreads();
    for (int k = 128; k > 0; k >>= 1) {
        if (tid < k) red[tid] += red[tid + k];
        __syncthreads();
    }
    if (tid == 0) *out = red[0];
}

__device__ inline float rbf5(float x, float a0)
{
    float s = 0.f, a = a0;
    #pragma unroll
    for (int m = 0; m < 5; ++m) { s += __expf(-x * a); a *= 0.5f; }
    return s;
}

// ---------------------------------------------------------------------------
// Kernel 5: weighted multi-kernel loss over the three Gram quadrants.
// Block = 32x32 (i,j) tile; deterministic per-block tree reduce.
// ---------------------------------------------------------------------------
__global__ __launch_bounds__(256)
void loss_kernel(const float* __restrict__ gram, const float* __restrict__ sq,
                 const float* __restrict__ src_label, const float* __restrict__ tgt_logits,
                 const float* __restrict__ sscale, const float* __restrict__ tscale,
                 const float* __restrict__ sumL2, float* __restrict__ losspart)
{
    __shared__ float snI[32][12], snJ[32][12], tnI[32][12], tnJ[32][12];
    __shared__ float red[256];
    const int tid = threadIdx.x;
    const int i0 = blockIdx.y * 32, j0 = blockIdx.x * 32;

    for (int idx = tid; idx < 32 * 12; idx += 256) {
        const int r = idx / 12, c = idx % 12;
        snI[r][c] = src_label[(size_t)(i0 + r) * NC + c] * sscale[c];
        snJ[r][c] = src_label[(size_t)(j0 + r) * NC + c] * sscale[c];
        tnI[r][c] = tgt_logits[(size_t)(i0 + r) * NC + c] * tscale[c];
        tnJ[r][c] = tgt_logits[(size_t)(j0 + r) * NC + c] * tscale[c];
    }
    __syncthreads();

    // bandwidth = sumL2/(n*n - n) / KERNEL_MUL^(KERNEL_NUM/2) ; a0 = 1/bandwidth
    const float S  = *sumL2;
    const float a0 = (16773120.0f * 4.0f) / S;

    const int di = tid >> 3;        // 0..31
    const int i  = i0 + di;
    const float sq_si = sq[i], sq_ti = sq[i + BD];
    const float* g1row = gram + (size_t)i * NN;               // [:b,:b]
    const float* g2row = gram + (size_t)(i + BD) * NN + BD;   // [b:,b:]
    const float* g3row = g1row + BD;                          // [:b,b:]

    float part = 0.f;
    #pragma unroll
    for (int k = 0; k < 4; ++k) {
        const int dj = (tid & 7) + 8 * k;
        const int j  = j0 + dj;
        const float g1 = g1row[j], g2 = g2row[j], g3 = g3row[j];
        const float sq_sj = sq[j], sq_tj = sq[j + BD];
        const float l2ss = fmaxf(sq_si + sq_sj - 2.f * g1, 0.f);
        const float l2tt = fmaxf(sq_ti + sq_tj - 2.f * g2, 0.f);
        const float l2st = fmaxf(sq_si + sq_tj - 2.f * g3, 0.f);
        const float kss = rbf5(l2ss, a0);
        const float ktt = rbf5(l2tt, a0);
        const float kst = rbf5(l2st, a0);
        float wss = 0.f, wtt = 0.f, wst = 0.f;
        #pragma unroll
        for (int c = 0; c < NC; ++c) {
            wss += snI[di][c] * snJ[dj][c];
            wtt += tnI[di][c] * tnJ[dj][c];
            wst += snI[di][c] * tnJ[dj][c];
        }
        part += wss * kss + wtt * ktt - 2.f * wst * kst;
    }
    part *= (1.f / 12.f);

    red[tid] = part;
    __syncthreads();
    for (int k = 128; k > 0; k >>= 1) {
        if (tid < k) red[tid] += red[tid + k];
        __syncthreads();
    }
    if (tid == 0) losspart[blockIdx.y * 64 + blockIdx.x] = red[0];
}

// ---------------------------------------------------------------------------
extern "C" void kernel_launch(void* const* d_in, const int* in_sizes, int n_in,
                              void* d_out, int out_size, void* d_ws, size_t ws_size,
                              hipStream_t stream)
{
    const float* source     = (const float*)d_in[0];
    const float* target     = (const float*)d_in[1];
    const float* src_label  = (const float*)d_in[2];
    const float* tgt_logits = (const float*)d_in[3];

    char* ws = (char*)d_ws;
    __bf16* Thi  = (__bf16*)ws;                                      //  8 MB
    __bf16* Tlo  = (__bf16*)(ws + (size_t)8 * 1024 * 1024);          //  8 MB
    float*  gram = (float*)(ws + (size_t)16 * 1024 * 1024);          // 64 MB
    char*   tail = ws + (size_t)80 * 1024 * 1024;
    float*  sq       = (float*)tail;                  // 4096 floats
    float*  sscale   = (float*)(tail + 16384);        // 12
    float*  tscale   = sscale + 12;                   // 12
    float*  sumL2    = tscale + 12;                   // 1
    float*  l2part   = (float*)(tail + 20480);        // 1024 used
    float*  losspart = (float*)(tail + 36864);        // 4096

    prep_kernel<<<1, 256, 0, stream>>>(src_label, tgt_logits, sscale, tscale);
    rows_kernel<<<NN, 256, 0, stream>>>(source, target, Thi, Tlo, sq);
    gram_kernel<<<dim3(32, 32), 128, 0, stream>>>(Thi, Tlo, sq, gram, l2part);
    reduce_kernel<<<1, 256, 0, stream>>>(l2part, 1024, sumL2);
    loss_kernel<<<dim3(64, 64), 256, 0, stream>>>(gram, sq, src_label, tgt_logits,
                                                  sscale, tscale, sumL2, losspart);
    reduce_kernel<<<1, 256, 0, stream>>>(losspart, 4096, (float*)d_out);
}
